// Attention_5214090297660
// MI455X (gfx1250) — matrix operand order repaired and checked by reading, not compiled, not run on hardware
//
#include <hip/hip_runtime.h>
#include <hip/hip_bf16.h>

// ---------------------------------------------------------------------------
// CDNA5 (gfx1250) fused causal multi-head attention block.
//   qkv  = x @ W_attn + b_attn              (WMMA f16 GEMM, f32 accum)
//   attn = flash-softmax(Q K^T / 8) V       (WMMA f16, streaming softmax)
//   out  = attn @ W_proj + b_proj           (WMMA f16 GEMM, f32 out)
// Matrix math: v_wmma_f32_16x16x32_f16 (wave32, 8 waves/block).
// Data movement: TDM tensor_load_to_lds (Q tiles), global async->LDS copies
// (K tiles, GEMM A tiles), global_prefetch for next tiles.
// ---------------------------------------------------------------------------

typedef __attribute__((ext_vector_type(16))) _Float16 v16h;
typedef __attribute__((ext_vector_type(8)))  _Float16 v8h;
typedef __attribute__((ext_vector_type(4)))  _Float16 v4h;
typedef __attribute__((ext_vector_type(8)))  float    v8f;
typedef __attribute__((ext_vector_type(4)))  unsigned v4u;
typedef __attribute__((ext_vector_type(8)))  int      v8i;
typedef __attribute__((ext_vector_type(4)))  int      v4i;

static constexpr int BATCH = 2;
static constexpr int TSEQ  = 2048;
static constexpr int DMODEL = 1024;
static constexpr int NHEAD = 16;
static constexpr int DHEAD = 64;       // DMODEL / NHEAD
static constexpr int D3    = 3 * DMODEL;
static constexpr int MROWS = BATCH * TSEQ;   // 4096

#define AS1 __attribute__((address_space(1)))
#define AS3 __attribute__((address_space(3)))

__device__ __forceinline__ v8f wmma_f16(v16h a, v16h b, v8f c) {
  return __builtin_amdgcn_wmma_f32_16x16x32_f16(
      /*neg_a=*/false, a, /*neg_b=*/false, b,
      /*c_mod=*/(short)0, c, /*reuse_a=*/false, /*reuse_b=*/false);
}

// Low 32 bits of a generic LDS pointer == LDS byte offset (ISA 10.2 aperture).
__device__ __forceinline__ unsigned lds_off(const void* p) {
  return (unsigned)(unsigned long long)p;
}

// ---- CDNA5 async global->LDS copy (per-lane 16B), ASYNCcnt tracked --------
__device__ __forceinline__ void async_load_b128(const void* gp, void* lp) {
#if __has_builtin(__builtin_amdgcn_global_load_async_to_lds_b128)
  __builtin_amdgcn_global_load_async_to_lds_b128(
      (AS1 v4i*)(unsigned long long)gp,
      (AS3 v4i*)lds_off(lp), 0, 0);
#else
  unsigned l = lds_off(lp);
  unsigned long long g = (unsigned long long)gp;
  asm volatile("global_load_async_to_lds_b128 %0, %1, off"
               :: "v"(l), "v"(g) : "memory");
#endif
}

__device__ __forceinline__ void wait_async0() {
#if __has_builtin(__builtin_amdgcn_s_wait_asynccnt)
  __builtin_amdgcn_s_wait_asynccnt(0);
#else
  asm volatile("s_wait_asynccnt 0x0" ::: "memory");
#endif
}

__device__ __forceinline__ void wait_tensor0() {
#if __has_builtin(__builtin_amdgcn_s_wait_tensorcnt)
  __builtin_amdgcn_s_wait_tensorcnt(0);
#else
  asm volatile("s_wait_tensorcnt 0x0" ::: "memory");
#endif
}

// ---- TDM: 2D f16 tile (tile_d0 x tile_d1), row stride stride0 elements, ---
// ---- LDS rows padded: pad_int_code DWORD interval, pad_amt_code DWORDs ----
#if __has_builtin(__builtin_amdgcn_tensor_load_to_lds)
#define HAVE_TDM 1
__device__ __forceinline__ void tdm_load_tile_f16(
    unsigned lds_addr, unsigned long long gaddr,
    unsigned tile_d0, unsigned tile_d1, unsigned stride0,
    unsigned pad_int_code, unsigned pad_amt_code) {
  // D# group 0: count=1, lds_addr, global_addr[56:0], type=2
  v4u g0;
  g0[0] = 1u;
  g0[1] = lds_addr;
  g0[2] = (unsigned)(gaddr & 0xffffffffu);
  g0[3] = (unsigned)((gaddr >> 32) & 0x01ffffffu) | (2u << 30);
  // D# group 1: data_size=2B, pad_enable, tensor dims huge (no OOB), tile dims
  v8i g1;
  g1[0] = (int)((1u << 16) | (1u << 20) | (pad_int_code << 22) |
                (pad_amt_code << 25));
  g1[1] = (int)0xffff0000u;                   // tensor_dim0[15:0]=0xffff
  g1[2] = (int)(0x7fffu | 0xffff0000u);       // td0[31:16]=0x7fff, td1[15:0]
  g1[3] = (int)(0x7fffu | (tile_d0 << 16));   // td1[31:16], tile_dim0
  g1[4] = (int)tile_d1;                       // tile_dim1, tile_dim2=0
  g1[5] = (int)stride0;                       // tensor_dim0_stride[31:0]
  g1[6] = 0;
  g1[7] = 0;
  v4i z4; z4[0] = 0; z4[1] = 0; z4[2] = 0; z4[3] = 0;
  v8i z8;
#pragma unroll
  for (int i = 0; i < 8; ++i) z8[i] = 0;
  __builtin_amdgcn_tensor_load_to_lds(g0, g1, z4, z4, z8, 0);
}
#else
#define HAVE_TDM 0
#endif

// A fragment (16x32, MxK) from row-major LDS tile; ld in halves (multiple of 8).
__device__ __forceinline__ v16h frag_a16(const _Float16* base, int ld, int lane) {
  const _Float16* p = base + (lane & 15) * ld + ((lane >> 4) << 3);
  v8h lo = *(const v8h*)p;
  v8h hi = *(const v8h*)(p + 16);
  v16h f;
#pragma unroll
  for (int i = 0; i < 8; ++i) { f[i] = lo[i]; f[i + 8] = hi[i]; }
  return f;
}

// B fragment (32x16, KxN) from N-major (transposed) LDS tile.
__device__ __forceinline__ v16h frag_b16(const _Float16* base, int ld, int lane) {
  const _Float16* p = base + (lane & 15) * ld + ((lane >> 4) << 3);
  v8h lo = *(const v8h*)p;
  v8h hi = *(const v8h*)(p + 16);
  v16h f;
#pragma unroll
  for (int i = 0; i < 8; ++i) { f[i] = lo[i]; f[i + 8] = hi[i]; }
  return f;
}

// ---------------------------------------------------------------------------
// fp32 -> fp16 convert (4 elems / thread)
// ---------------------------------------------------------------------------
__global__ void f32_to_f16_kernel(const float* __restrict__ in,
                                  _Float16* __restrict__ out, int n4) {
  int i = blockIdx.x * blockDim.x + threadIdx.x;
  if (i < n4) {
    float4 v = ((const float4*)in)[i];
    v4h o;
    o[0] = (_Float16)v.x; o[1] = (_Float16)v.y;
    o[2] = (_Float16)v.z; o[3] = (_Float16)v.w;
    ((v4h*)out)[i] = o;
  }
}

// ---------------------------------------------------------------------------
// WMMA GEMM: C[M,N] = A[M,K] * B[K,N] + bias[N]
// Block tile 128x128, 8 waves (2x4), wave tile 64x32, K-step 32.
// A tile staged via async global->LDS; B tile transposed through VGPRs.
// ---------------------------------------------------------------------------
template <bool HALF_OUT>
__global__ __launch_bounds__(256) void gemm_f16_kernel(
    const _Float16* __restrict__ A, const _Float16* __restrict__ B,
    const float* __restrict__ bias, void* __restrict__ Cout,
    int M, int N, int K) {
  __shared__ alignas(16) _Float16 As[128 * 40];  // row-major [m][k], pad 8
  __shared__ alignas(16) _Float16 Bs[128 * 40];  // transposed [n][k], pad 8

  const int tid = threadIdx.x;
  const int lane = tid & 31;
  const int wave = tid >> 5;
  const int wm = wave >> 2;
  const int wn = wave & 3;
  const int bm = blockIdx.y * 128;
  const int bn = blockIdx.x * 128;

  const v8f vzero = {0.f, 0.f, 0.f, 0.f, 0.f, 0.f, 0.f, 0.f};
  v8f acc[4][2];
#pragma unroll
  for (int mi = 0; mi < 4; ++mi)
#pragma unroll
    for (int ni = 0; ni < 2; ++ni) acc[mi][ni] = vzero;

  for (int k0 = 0; k0 < K; k0 += 32) {
    // A tile: 128 x 32 halves -> async DMA into LDS (no VGPR round trip)
    for (int c = tid; c < 512; c += 256) {
      int r = c >> 2, cc = (c & 3) << 3;
      async_load_b128(A + (size_t)(bm + r) * K + k0 + cc, &As[r * 40 + cc]);
    }
    // B tile transposed into Bs[n][k]: 32 x 128 halves
    for (int c = tid; c < 512; c += 256) {
      int kk = c >> 4, nn = (c & 15) << 3;
      v8h t = *(const v8h*)(B + (size_t)(k0 + kk) * N + bn + nn);
#pragma unroll
      for (int e = 0; e < 8; ++e) Bs[(nn + e) * 40 + kk] = t[e];
      if (k0 + 32 < K)
        __builtin_prefetch(B + (size_t)(k0 + 32 + kk) * N + bn + nn, 0, 2);
    }
    wait_async0();
    __syncthreads();

    v16h af[4], bf[2];
#pragma unroll
    for (int mi = 0; mi < 4; ++mi)
      af[mi] = frag_a16(As + (wm * 64 + mi * 16) * 40, 40, lane);
#pragma unroll
    for (int ni = 0; ni < 2; ++ni)
      bf[ni] = frag_b16(Bs + (wn * 32 + ni * 16) * 40, 40, lane);
#pragma unroll
    for (int mi = 0; mi < 4; ++mi)
#pragma unroll
      for (int ni = 0; ni < 2; ++ni)
        acc[mi][ni] = wmma_f16(af[mi], bf[ni], acc[mi][ni]);
    __syncthreads();
  }

#pragma unroll
  for (int mi = 0; mi < 4; ++mi)
#pragma unroll
    for (int ni = 0; ni < 2; ++ni) {
      const int col = bn + wn * 32 + ni * 16 + (lane & 15);
      const float bv = bias[col];
#pragma unroll
      for (int e = 0; e < 8; ++e) {
        const int row = bm + wm * 64 + mi * 16 + e + ((lane >> 4) << 3);
        const float v = acc[mi][ni][e] + bv;
        if (HALF_OUT)
          ((_Float16*)Cout)[(size_t)row * N + col] = (_Float16)v;
        else
          ((float*)Cout)[(size_t)row * N + col] = v;
      }
    }
}

// ---------------------------------------------------------------------------
// Flash-style causal attention, one (b, h, 64-row Q block) per 256-thread block.
// Q tile via TDM (padded 2D tile DMA); K tiles via async copies; V transposed
// through VGPRs.  S = QK^T*0.125 via WMMA; online softmax in LDS (P stored f16
// in place over the f32 score buffer); O += P V via WMMA; final 1/l normalize.
// ---------------------------------------------------------------------------
__global__ __launch_bounds__(256) void attn_kernel(
    const _Float16* __restrict__ qkv,  // [MROWS][3072] f16
    _Float16* __restrict__ Ah) {       // [MROWS][1024] f16 (heads merged)
  __shared__ alignas(16) _Float16 Qs[64 * 72];   // [q][d], pad 8
  __shared__ alignas(16) _Float16 Ks[64 * 72];   // [kidx][d], pad 8
  __shared__ alignas(16) _Float16 Vts[64 * 72];  // [d][kidx], pad 8 (transposed)
  __shared__ alignas(16) float Ssc[64 * 68];     // scores [q][kidx], pad 4
  __shared__ float Mrow[64], Lrow[64], Arow[64];

  const int tid = threadIdx.x;
  const int lane = tid & 31;
  const int wave = tid >> 5;
  const int wm = wave >> 1;   // 0..3 -> 16-row slab (S and O phases)
  const int wn = wave & 1;    // 0..1 -> 32-col slab
  const int qb = blockIdx.x;  // 0..31
  const int h  = blockIdx.y;  // 0..15
  const int b  = blockIdx.z;  // 0..1

  const size_t tok0 = (size_t)b * TSEQ + (size_t)qb * 64;
  const int qoff = h * DHEAD;
  const int koff = DMODEL + h * DHEAD;
  const int voff = 2 * DMODEL + h * DHEAD;

  // ---- Q tile: 64x64 f16, row stride 3072 elems, LDS rows padded to 72 ----
#if HAVE_TDM
  if (wave == 0) {
    // pad_interval: 32 DWORDs (=128B row) -> code 4; pad: 4 DWORDs -> code 3
    tdm_load_tile_f16(lds_off(Qs),
                      (unsigned long long)(qkv + tok0 * D3 + qoff),
                      /*tile_d0=*/64, /*tile_d1=*/64, /*stride0=*/D3,
                      /*pad_int=*/4, /*pad_amt=*/3);
    wait_tensor0();
  }
#else
  for (int c = tid; c < 512; c += 256) {
    int r = c >> 3, cc = (c & 7) << 3;
    async_load_b128(qkv + (tok0 + r) * D3 + qoff + cc, &Qs[r * 72 + cc]);
  }
  wait_async0();
#endif
  if (tid < 64) { Mrow[tid] = -3.0e38f; Lrow[tid] = 0.0f; }

  const v8f vzero = {0.f, 0.f, 0.f, 0.f, 0.f, 0.f, 0.f, 0.f};
  v8f o[2];
  o[0] = vzero; o[1] = vzero;
  __syncthreads();

  for (int j = 0; j <= qb; ++j) {
    const size_t ktok0 = (size_t)b * TSEQ + (size_t)j * 64;
    // K tile via async DMA; V tile transposed through VGPRs; prefetch next j.
    for (int c = tid; c < 512; c += 256) {
      int r = c >> 3, cc = (c & 7) << 3;
      async_load_b128(qkv + (ktok0 + r) * D3 + koff + cc, &Ks[r * 72 + cc]);
      v8h t = *(const v8h*)(qkv + (ktok0 + r) * D3 + voff + cc);
#pragma unroll
      for (int e = 0; e < 8; ++e) Vts[(cc + e) * 72 + r] = t[e];
      if (j < qb) {
        __builtin_prefetch(qkv + (ktok0 + 64 + r) * D3 + koff + cc, 0, 2);
        __builtin_prefetch(qkv + (ktok0 + 64 + r) * D3 + voff + cc, 0, 2);
      }
    }
    wait_async0();
    __syncthreads();

    // ---- S = Q K^T (wave tile 16x32, reduction over dh=64) ----
    v8f s[2];
    s[0] = vzero; s[1] = vzero;
#pragma unroll
    for (int d0 = 0; d0 < 64; d0 += 32) {
      v16h qf = frag_a16(Qs + wm * 16 * 72 + d0, 72, lane);
#pragma unroll
      for (int ni = 0; ni < 2; ++ni) {
        v16h kf = frag_b16(Ks + (wn * 32 + ni * 16) * 72 + d0, 72, lane);
        s[ni] = wmma_f16(qf, kf, s[ni]);
      }
    }
#pragma unroll
    for (int ni = 0; ni < 2; ++ni)
#pragma unroll
      for (int e = 0; e < 8; ++e) {
        const int row = wm * 16 + e + ((lane >> 4) << 3);
        const int col = wn * 32 + ni * 16 + (lane & 15);
        Ssc[row * 68 + col] = s[ni][e] * 0.125f;
      }
    __syncthreads();

    // ---- Online softmax row pass (thread r owns query row r) ----
    if (tid < 64) {
      const int r = tid;
      const int gq = qb * 64 + r;
      int kmax = gq - j * 64 + 1;      // causal: keys <= query
      if (kmax > 64) kmax = 64;
      float* Srow = Ssc + r * 68;
      float mj = -3.0e38f;
      for (int k = 0; k < kmax; ++k) mj = fmaxf(mj, Srow[k]);
      const float mold = Mrow[r];
      const float mnew = fmaxf(mold, mj);
      const float alpha = __expf(mold - mnew);
      _Float16* Pr = (_Float16*)Srow;  // f16 P overwrites f32 S in place
      float sum = 0.0f;
      for (int k = 0; k < 64; ++k) {
        const float p = (k < kmax) ? __expf(Srow[k] - mnew) : 0.0f;
        sum += p;
        Pr[k] = (_Float16)p;
      }
      Mrow[r] = mnew;
      Lrow[r] = Lrow[r] * alpha + sum;
      Arow[r] = alpha;
    }
    __syncthreads();

    // ---- Rescale O, then O += P V (wave tile 16x32 over [q][d]) ----
#pragma unroll
    for (int ni = 0; ni < 2; ++ni)
#pragma unroll
      for (int e = 0; e < 8; ++e) {
        const int row = wm * 16 + e + ((lane >> 4) << 3);
        o[ni][e] *= Arow[row];
      }
    const _Float16* Ps = (const _Float16*)Ssc;  // row stride 136 halves
#pragma unroll
    for (int c0 = 0; c0 < 64; c0 += 32) {
      v16h pf = frag_a16(Ps + wm * 16 * 136 + c0, 136, lane);
#pragma unroll
      for (int ni = 0; ni < 2; ++ni) {
        v16h vf = frag_b16(Vts + (wn * 32 + ni * 16) * 72 + c0, 72, lane);
        o[ni] = wmma_f16(pf, vf, o[ni]);
      }
    }
    __syncthreads();  // protect Ks/Vts/Ssc before next j
  }

  // ---- Normalize and store (merged-head layout) ----
#pragma unroll
  for (int ni = 0; ni < 2; ++ni)
#pragma unroll
    for (int e = 0; e < 8; ++e) {
      const int row = wm * 16 + e + ((lane >> 4) << 3);
      const int col = wn * 32 + ni * 16 + (lane & 15);
      const float val = o[ni][e] / Lrow[row];
      Ah[(tok0 + row) * DMODEL + qoff + col] = (_Float16)val;
    }
}

// ---------------------------------------------------------------------------
// Launch: convert -> QKV GEMM -> attention -> proj GEMM
// ---------------------------------------------------------------------------
extern "C" void kernel_launch(void* const* d_in, const int* in_sizes, int n_in,
                              void* d_out, int out_size, void* d_ws, size_t ws_size,
                              hipStream_t stream) {
  const float* x      = (const float*)d_in[0];  // [2,2048,1024]
  const float* W_attn = (const float*)d_in[1];  // [1024,3072]
  const float* b_attn = (const float*)d_in[2];  // [3072]
  const float* W_proj = (const float*)d_in[3];  // [1024,1024]
  const float* b_proj = (const float*)d_in[4];  // [1024]

  char* ws = (char*)d_ws;
  const size_t xh_off   = 0;                                    // 8 MB
  const size_t wah_off  = xh_off  + (size_t)MROWS * DMODEL * 2; // 6 MB
  const size_t wph_off  = wah_off + (size_t)DMODEL * D3 * 2;    // 2 MB
  const size_t qkv_off  = wph_off + (size_t)DMODEL * DMODEL * 2;// 24 MB
  const size_t ah_off   = qkv_off + (size_t)MROWS * D3 * 2;     // 8 MB

  _Float16* Xh   = (_Float16*)(ws + xh_off);
  _Float16* Wah  = (_Float16*)(ws + wah_off);
  _Float16* Wph  = (_Float16*)(ws + wph_off);
  _Float16* qkvh = (_Float16*)(ws + qkv_off);
  _Float16* Ah   = (_Float16*)(ws + ah_off);

  {
    int n4 = MROWS * DMODEL / 4;
    f32_to_f16_kernel<<<(n4 + 255) / 256, 256, 0, stream>>>(x, Xh, n4);
    n4 = DMODEL * D3 / 4;
    f32_to_f16_kernel<<<(n4 + 255) / 256, 256, 0, stream>>>(W_attn, Wah, n4);
    n4 = DMODEL * DMODEL / 4;
    f32_to_f16_kernel<<<(n4 + 255) / 256, 256, 0, stream>>>(W_proj, Wph, n4);
  }

  // qkv = Xh @ Wah + b_attn   (M=4096, N=3072, K=1024) -> f16
  gemm_f16_kernel<true><<<dim3(D3 / 128, MROWS / 128), 256, 0, stream>>>(
      Xh, Wah, b_attn, (void*)qkvh, MROWS, D3, DMODEL);

  // causal attention -> Ah f16 [4096, 1024]
  attn_kernel<<<dim3(TSEQ / 64, NHEAD, BATCH), 256, 0, stream>>>(qkvh, Ah);

  // out = Ah @ Wph + b_proj   (M=4096, N=1024, K=1024) -> f32
  gemm_f16_kernel<false><<<dim3(DMODEL / 128, MROWS / 128), 256, 0, stream>>>(
      Ah, Wph, b_proj, d_out, MROWS, DMODEL, DMODEL);
}
